// SeqOnlyTransformer_83760452206848
// MI455X (gfx1250) — compile-verified, matrix operand order checked by reading, not run on hardware
//
#include <hip/hip_runtime.h>

// Problem constants
static constexpr int cB = 4, cL = 256, cA = 48, cH = 256, cNH = 4, cNL = 4, cNB = 64, cDH = 64;
static constexpr int BL = cB * cL;            // 1024 token rows
#define EPSf 1e-5f

typedef _Float16 f16;
typedef __attribute__((ext_vector_type(16))) f16   wh16;
typedef __attribute__((ext_vector_type(8)))  f16   wh8;
typedef __attribute__((ext_vector_type(8)))  float wf8;

// ---- WMMA fragment helpers (CDNA5 wave32 layouts) -------------------------
// A (16x32 f16): lane%16 = row, elements 0..7 -> K = 8*(lane/16)+e,
//                elements 8..15 -> K = 16 + 8*(lane/16) + (e-8).
// B stored transposed [N][K]: identical per-lane pattern with lane%16 = col.
__device__ __forceinline__ wh16 frag_ld(const f16* p) {
  wh8 lo = *(const wh8*)(p);
  wh8 hi = *(const wh8*)(p + 16);
  wh16 r;
#pragma unroll
  for (int i = 0; i < 8; ++i) { r[i] = lo[i]; r[i + 8] = hi[i]; }
  return r;
}

__device__ __forceinline__ wf8 wmma32(wh16 a, wh16 b, wf8 c) {
  return __builtin_amdgcn_wmma_f32_16x16x32_f16(false, a, false, b, (short)0, c, false, false);
}

// ---- weight convert/transpose: Wt[n*K+k] = (f16) W[k*N+n] -----------------
__global__ __launch_bounds__(256) void convt_kernel(const float* W, f16* Wt, int K, int N) {
  int t = blockIdx.x * 256 + threadIdx.x;
  if (t >= K * N) return;
  int k = t % K, n = t / K;
  Wt[t] = (f16)W[(long)k * N + n];
}

// pw1 halves with pln_w folded in: Wa[n*256+k] = pw1[k][n]*pln_w[k], Wb for k+256
__global__ __launch_bounds__(256) void pw1ab_kernel(const float* pw1, const float* plnw,
                                                    f16* Wa, f16* Wb) {
  int t = blockIdx.x * 256 + threadIdx.x;   // 65536
  int k = t & 255, n = t >> 8;
  Wa[t] = (f16)(pw1[k * 256 + n] * plnw[k]);
  Wb[t] = (f16)(pw1[(k + 256) * 256 + n] * plnw[k + 256]);
}

__global__ __launch_bounds__(256) void pw2t_kernel(const float* pw2, f16* Wt) {
  int t = blockIdx.x * 256 + threadIdx.x;   // 16384 : [n<64][k<256]
  int k = t & 255, n = t >> 8;
  Wt[t] = (f16)pw2[k * 64 + n];
}

// z[n] = sum_k pln_w[k]*pw1[k][n] ; c0[n] = pb1[n] + sum_k pln_b[k]*pw1[k][n]
__global__ __launch_bounds__(256) void zc0_kernel(const float* pw1, const float* plnw,
                                                  const float* plnb, const float* pb1,
                                                  float* z, float* c0) {
  int n = threadIdx.x;
  float zz = 0.f, cc = pb1[n];
  for (int k = 0; k < 512; ++k) {
    float w = pw1[k * 256 + n];
    zz += plnw[k] * w;
    cc += plnb[k] * w;
  }
  z[n] = zz; c0[n] = cc;
}

// ---- embed: h = x @ in_w + in_b + pos -------------------------------------
__global__ __launch_bounds__(256) void embed_kernel(const float* x, const float* in_w,
                                                    const float* in_b, const float* pos,
                                                    float* h) {
  int row = blockIdx.x;        // b*L + l
  int col = threadIdx.x;
  int l = row & (cL - 1);
  float acc = in_b[col] + pos[l * cH + col];
  const float* xr = x + row * cA;
  for (int a = 0; a < cA; ++a) acc += xr[a] * in_w[a * cH + col];
  h[row * cH + col] = acc;
}

// ---- LayerNorm over H=256, one block per row, f16 output ------------------
__global__ __launch_bounds__(256) void ln_kernel(const float* X, const float* w,
                                                 const float* bb, f16* Y) {
  __shared__ float red[256];
  int row = blockIdx.x, t = threadIdx.x;
  float x = X[row * 256 + t];
  red[t] = x; __syncthreads();
  for (int s = 128; s > 0; s >>= 1) { if (t < s) red[t] += red[t + s]; __syncthreads(); }
  float mu = red[0] * (1.0f / 256.0f);
  __syncthreads();
  float d = x - mu;
  red[t] = d * d; __syncthreads();
  for (int s = 128; s > 0; s >>= 1) { if (t < s) red[t] += red[t + s]; __syncthreads(); }
  float var = red[0] * (1.0f / 256.0f);
  Y[row * 256 + t] = (f16)(d * rsqrtf(var + EPSf) * w[t] + bb[t]);
}

// ---- per-token stats for pair head ----------------------------------------
__global__ __launch_bounds__(256) void stats_kernel(const float* X, float* mu, float* m2) {
  __shared__ float r1[256];
  __shared__ float r2[256];
  int row = blockIdx.x, t = threadIdx.x;
  float x = X[row * 256 + t];
  r1[t] = x; r2[t] = x * x; __syncthreads();
  for (int s = 128; s > 0; s >>= 1) {
    if (t < s) { r1[t] += r1[t + s]; r2[t] += r2[t + s]; }
    __syncthreads();
  }
  if (t == 0) { mu[row] = r1[0] * (1.0f / 256.0f); m2[row] = r2[0] * (1.0f / 256.0f); }
}

__global__ __launch_bounds__(256) void tofp16_kernel(const float* X, f16* Y, int n) {
  int t = blockIdx.x * 256 + threadIdx.x;
  if (t < n) Y[t] = (f16)X[t];
}

// ---- generic WMMA GEMM: C(MxN) = A(MxK,f16) @ Bt(NxK,f16) + epilogue ------
// block = 256 thr = 8 waves; block tile 128x64; wave tile 32x32 (2x2 frags)
__global__ __launch_bounds__(256) void gemm_kernel(const f16* __restrict__ A,
                                                   const f16* __restrict__ Bt,
                                                   const float* __restrict__ bias,
                                                   const float* __restrict__ resid,
                                                   float* Cf, f16* Ch,
                                                   int N, int K, int relu) {
  int wave = threadIdx.x >> 5, lane = threadIdx.x & 31;
  int idx = lane & 15, koff = (lane >> 4) * 8;
  int m_base = blockIdx.x * 128 + (wave >> 1) * 32;
  int n_base = blockIdx.y * 64 + (wave & 1) * 32;
  wf8 acc[2][2] = {};
  for (int k0 = 0; k0 < K; k0 += 32) {
    wh16 a0 = frag_ld(A + (long)(m_base + idx) * K + k0 + koff);
    wh16 a1 = frag_ld(A + (long)(m_base + 16 + idx) * K + k0 + koff);
    wh16 b0 = frag_ld(Bt + (long)(n_base + idx) * K + k0 + koff);
    wh16 b1 = frag_ld(Bt + (long)(n_base + 16 + idx) * K + k0 + koff);
    acc[0][0] = wmma32(a0, b0, acc[0][0]);
    acc[0][1] = wmma32(a0, b1, acc[0][1]);
    acc[1][0] = wmma32(a1, b0, acc[1][0]);
    acc[1][1] = wmma32(a1, b1, acc[1][1]);
  }
#pragma unroll
  for (int im = 0; im < 2; ++im)
#pragma unroll
    for (int in = 0; in < 2; ++in)
#pragma unroll
      for (int v = 0; v < 8; ++v) {
        int row = m_base + im * 16 + v + 8 * (lane >> 4);
        int col = n_base + in * 16 + idx;
        float val = acc[im][in][v];
        if (bias)  val += bias[col];
        if (resid) val += resid[(long)row * N + col];
        if (relu)  val = fmaxf(val, 0.0f);
        long o = (long)row * N + col;
        if (Cf) Cf[o] = val;
        if (Ch) Ch[o] = (f16)val;
      }
}

// ---- fused attention: one block per (b, head, 32-query tile) --------------
__global__ __launch_bounds__(256) void attn_kernel(const f16* __restrict__ qkv, f16* o16) {
  __shared__ __align__(16) float S[32][256];
  __shared__ __align__(16) f16   P[32][256];
  __shared__ float red[32][8];
  __shared__ float rowval[32];
  int blk = blockIdx.x;                 // b*32 + hd*8 + qt
  int qt = blk & 7;
  int hd = (blk >> 3) & 3;
  int b  = blk >> 5;
  int q0 = qt * 32;
  int tid = threadIdx.x;
  int wave = tid >> 5, lane = tid & 31;
  int idx = lane & 15, koff = (lane >> 4) * 8;

  // phase 1: S = (Q @ K^T) * 1/sqrt(DH)
  {
    int mt = wave & 1;
    int ktg = wave >> 1;                // 0..3, 4 key tiles each
    for (int i = 0; i < 4; ++i) {
      int kt = ktg * 4 + i;
      wf8 acc = {};
      for (int ks = 0; ks < 2; ++ks) {
        int k0 = ks * 32;
        const f16* ap = qkv + (long)(b * cL + q0 + mt * 16 + idx) * 768 + hd * 64 + k0 + koff;
        const f16* bp = qkv + (long)(b * cL + kt * 16 + idx) * 768 + 256 + hd * 64 + k0 + koff;
        acc = wmma32(frag_ld(ap), frag_ld(bp), acc);
      }
#pragma unroll
      for (int v = 0; v < 8; ++v)
        S[mt * 16 + v + 8 * (lane >> 4)][kt * 16 + idx] = acc[v] * 0.125f;
    }
  }
  __syncthreads();

  // softmax over 256 keys (8 threads per row)
  {
    int row = tid >> 3, seg = tid & 7;
    float m = -1e30f;
    for (int k = seg * 32; k < seg * 32 + 32; ++k) m = fmaxf(m, S[row][k]);
    red[row][seg] = m; __syncthreads();
    if (seg == 0) {
      float mm = red[row][0];
      for (int q = 1; q < 8; ++q) mm = fmaxf(mm, red[row][q]);
      rowval[row] = mm;
    }
    __syncthreads();
    float mm = rowval[row];
    float sum = 0.f;
    for (int k = seg * 32; k < seg * 32 + 32; ++k) {
      float e = expf(S[row][k] - mm); S[row][k] = e; sum += e;
    }
    red[row][seg] = sum; __syncthreads();
    if (seg == 0) {
      float ss = 0.f;
      for (int q = 0; q < 8; ++q) ss += red[row][q];
      rowval[row] = ss;
    }
    __syncthreads();
    float inv = 1.0f / rowval[row];
    for (int k = seg * 32; k < seg * 32 + 32; ++k) P[row][k] = (f16)(S[row][k] * inv);
  }
  __syncthreads();

  // phase 2: O = P @ V  (B-frag gathered from v rows, K = key index)
  {
    int mt = wave & 1;
    int nt = wave >> 1;                 // 0..3
    wf8 acc = {};
    for (int ks = 0; ks < 8; ++ks) {
      int k0 = ks * 32;
      wh16 af = frag_ld(&P[mt * 16 + idx][k0 + koff]);
      wh16 bf;
#pragma unroll
      for (int e = 0; e < 8; ++e) {
        int key0 = k0 + koff + e;
        int key1 = k0 + 16 + koff + e;
        bf[e]     = qkv[(long)(b * cL + key0) * 768 + 512 + hd * 64 + nt * 16 + idx];
        bf[e + 8] = qkv[(long)(b * cL + key1) * 768 + 512 + hd * 64 + nt * 16 + idx];
      }
      acc = wmma32(af, bf, acc);
    }
#pragma unroll
    for (int v = 0; v < 8; ++v) {
      int qrow = q0 + mt * 16 + v + 8 * (lane >> 4);
      o16[(long)(b * cL + qrow) * 256 + hd * 64 + nt * 16 + idx] = (f16)acc[v];
    }
  }
}

// ---- pair head: block per (b,i) -------------------------------------------
// pw2t (64x256 f16 = 32KB) is staged into LDS once via CDNA5 async-to-LDS;
// gelu rows built 32 at a time with b128 LDS stores; WMMA GEMM vs LDS weights.
__global__ __launch_bounds__(256) void pair_kernel(const float* __restrict__ U,
                                                   const float* __restrict__ V,
                                                   const float* __restrict__ mu,
                                                   const float* __restrict__ m2,
                                                   const float* __restrict__ z,
                                                   const float* __restrict__ c0,
                                                   const float* __restrict__ pb2,
                                                   const f16* __restrict__ pw2t,
                                                   float* out) {
  __shared__ __align__(16) f16 Wlds[64 * 256];   // 32KB staged pw2t
  __shared__ __align__(16) f16 Ash[32][256];     // 16KB gelu tile
  __shared__ float Ui[256];
  __shared__ float zsh[256];
  __shared__ float c0sh[256];
  __shared__ float pb2sh[64];
  int bi = blockIdx.x;                  // b*L + i
  int b = bi >> 8;
  int tid = threadIdx.x;

  // async stage of pw2t -> LDS (2048 x 16B), overlapped with scalar setup
  {
    unsigned lbase = (unsigned)(size_t)(&Wlds[0]);
#pragma unroll
    for (int r = 0; r < 8; ++r) {
      int t = tid + r * 256;
      unsigned loff = lbase + t * 16;
      unsigned long long ga = (unsigned long long)(size_t)(pw2t + t * 8);
      asm volatile("global_load_async_to_lds_b128 %0, %1, off"
                   :: "v"(loff), "v"(ga) : "memory");
    }
  }
  Ui[tid] = U[bi * 256 + tid];
  zsh[tid] = z[tid];
  c0sh[tid] = c0[tid];
  if (tid < 64) pb2sh[tid] = pb2[tid];
  float mu_i = mu[bi], m2_i = m2[bi];
  asm volatile("s_wait_asynccnt 0" ::: "memory");
  __syncthreads();

  int wave = tid >> 5, lane = tid & 31;
  int idx = lane & 15, koff = (lane >> 4) * 8;

  for (int jc = 0; jc < 8; ++jc) {
    int j0 = jc * 32;
    // build 32 gelu'd pair rows: y = s*(U_i + V_j - mu_ij*z) + c0
    int jl = tid & 31;
    int cs = tid >> 5;                  // 8 col segments of 32
    int bj = (b << 8) + (j0 + jl);
    float muj = mu[bj], m2j = m2[bj];
    float mupair = 0.5f * (mu_i + muj);
    float var = 0.5f * (m2_i + m2j) - mupair * mupair;
    float s = rsqrtf(var + EPSf);
    const float4* Vj4 = (const float4*)(V + (long)bj * 256);
#pragma unroll
    for (int rr = 0; rr < 4; ++rr) {
      int n0 = cs * 32 + rr * 8;
      float4 va = Vj4[n0 / 4];
      float4 vb = Vj4[n0 / 4 + 1];
      float vv[8] = {va.x, va.y, va.z, va.w, vb.x, vb.y, vb.z, vb.w};
      wh8 g8;
#pragma unroll
      for (int e = 0; e < 8; ++e) {
        int n = n0 + e;
        float y = s * (Ui[n] + vv[e] - mupair * zsh[n]) + c0sh[n];
        g8[e] = (f16)(0.5f * y * (1.0f + erff(y * 0.70710678118654752f)));
      }
      *(wh8*)&Ash[jl][n0] = g8;
    }
    __syncthreads();
    // GEMM: (32 x 256) @ Wlds(64 x 256)^T -> 32 x 64 ; 1 tile per wave
    int mt = wave & 1;                  // 2 row tiles
    int nt = wave >> 1;                 // 4 col tiles
    wf8 acc = {};
    for (int ks = 0; ks < 8; ++ks) {
      int k0 = ks * 32;
      wh16 af = frag_ld(&Ash[mt * 16 + idx][k0 + koff]);
      wh16 bf = frag_ld(&Wlds[(nt * 16 + idx) * 256 + k0 + koff]);
      acc = wmma32(af, bf, acc);
    }
#pragma unroll
    for (int v = 0; v < 8; ++v) {
      int jrow = j0 + mt * 16 + v + 8 * (lane >> 4);
      int n = nt * 16 + idx;
      out[((long)bi * 256 + jrow) * 64 + n] = acc[v] + pb2sh[n];
    }
    __syncthreads();
  }
}

// ---- in-place symmetrize: out = (out + out^T(i<->j)) / 2, b128 traffic ----
__global__ __launch_bounds__(256) void sym_kernel(float* out) {
  long t = (long)blockIdx.x * 256 + threadIdx.x;   // over B*L*L*16 float4s
  int n4 = (int)(t & 15);
  long r = t >> 4;
  int j = (int)(r & 255);
  int i = (int)((r >> 8) & 255);
  int b = (int)(r >> 16);
  if (i < j) {
    float4* p1 = (float4*)(out + ((((long)(b * 256 + i)) * 256 + j) * 64 + n4 * 4));
    float4* p2 = (float4*)(out + ((((long)(b * 256 + j)) * 256 + i) * 64 + n4 * 4));
    float4 a = *p1, c = *p2;
    float4 m = make_float4(0.5f * (a.x + c.x), 0.5f * (a.y + c.y),
                           0.5f * (a.z + c.z), 0.5f * (a.w + c.w));
    *p1 = m; *p2 = m;
  }
}

extern "C" void kernel_launch(void* const* d_in, const int* in_sizes, int n_in,
                              void* d_out, int out_size, void* d_ws, size_t ws_size,
                              hipStream_t stream) {
  (void)in_sizes; (void)n_in; (void)out_size; (void)ws_size;
  const float* x     = (const float*)d_in[0];
  const float* in_w  = (const float*)d_in[1];
  const float* in_b  = (const float*)d_in[2];
  const float* pos   = (const float*)d_in[3];
  const float* ln1_w = (const float*)d_in[4];
  const float* ln1_b = (const float*)d_in[5];
  const float* qkv_w = (const float*)d_in[6];
  const float* qkv_b = (const float*)d_in[7];
  const float* out_w = (const float*)d_in[8];
  const float* out_b = (const float*)d_in[9];
  const float* ln2_w = (const float*)d_in[10];
  const float* ln2_b = (const float*)d_in[11];
  const float* ff1_w = (const float*)d_in[12];
  const float* ff1_b = (const float*)d_in[13];
  const float* ff2_w = (const float*)d_in[14];
  const float* ff2_b = (const float*)d_in[15];
  const float* pln_w = (const float*)d_in[16];
  const float* pln_b = (const float*)d_in[17];
  const float* pw1   = (const float*)d_in[18];
  const float* pb1   = (const float*)d_in[19];
  const float* pw2   = (const float*)d_in[20];
  const float* pb2   = (const float*)d_in[21];
  float* outp = (float*)d_out;

  char* ws = (char*)d_ws;
  size_t off = 0;
  auto alloc = [&](size_t bytes) -> void* {
    void* p = ws + off;
    off = (off + bytes + 255) & ~(size_t)255;
    return p;
  };

  float* h     = (float*)alloc((size_t)BL * cH * 4);
  f16*  a16    = (f16*)  alloc((size_t)BL * cH * 2);
  f16*  qkv16  = (f16*)  alloc((size_t)BL * 768 * 2);
  f16*  o16    = (f16*)  alloc((size_t)BL * cH * 2);
  f16*  ff16   = (f16*)  alloc((size_t)BL * 1024 * 2);
  f16*  h16    = (f16*)  alloc((size_t)BL * cH * 2);
  float* Ubuf  = (float*)alloc((size_t)BL * 256 * 4);
  float* Vbuf  = (float*)alloc((size_t)BL * 256 * 4);
  float* muv   = (float*)alloc((size_t)BL * 4);
  float* m2v   = (float*)alloc((size_t)BL * 4);
  float* zv    = (float*)alloc(256 * 4);
  float* c0v   = (float*)alloc(256 * 4);
  f16*  qkvwt  = (f16*)alloc((size_t)cNL * 768 * 256 * 2);
  f16*  outwt  = (f16*)alloc((size_t)cNL * 256 * 256 * 2);
  f16*  ff1wt  = (f16*)alloc((size_t)cNL * 1024 * 256 * 2);
  f16*  ff2wt  = (f16*)alloc((size_t)cNL * 256 * 1024 * 2);
  f16*  pw1a   = (f16*)alloc((size_t)256 * 256 * 2);
  f16*  pw1b   = (f16*)alloc((size_t)256 * 256 * 2);
  f16*  pw2t   = (f16*)alloc((size_t)64 * 256 * 2);

  // --- weight prep (f16, B-transposed [N][K]) ---
  for (int l = 0; l < cNL; ++l) {
    convt_kernel<<<(256 * 768 + 255) / 256, 256, 0, stream>>>(
        qkv_w + (size_t)l * 256 * 768, qkvwt + (size_t)l * 768 * 256, 256, 768);
    convt_kernel<<<(256 * 256 + 255) / 256, 256, 0, stream>>>(
        out_w + (size_t)l * 256 * 256, outwt + (size_t)l * 256 * 256, 256, 256);
    convt_kernel<<<(256 * 1024 + 255) / 256, 256, 0, stream>>>(
        ff1_w + (size_t)l * 256 * 1024, ff1wt + (size_t)l * 1024 * 256, 256, 1024);
    convt_kernel<<<(1024 * 256 + 255) / 256, 256, 0, stream>>>(
        ff2_w + (size_t)l * 1024 * 256, ff2wt + (size_t)l * 256 * 1024, 1024, 256);
  }
  pw1ab_kernel<<<256, 256, 0, stream>>>(pw1, pln_w, pw1a, pw1b);
  pw2t_kernel<<<64, 256, 0, stream>>>(pw2, pw2t);
  zc0_kernel<<<1, 256, 0, stream>>>(pw1, pln_w, pln_b, pb1, zv, c0v);

  // --- embed ---
  embed_kernel<<<BL, 256, 0, stream>>>(x, in_w, in_b, pos, h);

  // --- transformer layers ---
  for (int l = 0; l < cNL; ++l) {
    ln_kernel<<<BL, 256, 0, stream>>>(h, ln1_w + l * cH, ln1_b + l * cH, a16);
    gemm_kernel<<<dim3(BL / 128, 768 / 64), 256, 0, stream>>>(
        a16, qkvwt + (size_t)l * 768 * 256, qkv_b + (size_t)l * 768,
        nullptr, nullptr, qkv16, 768, 256, 0);
    attn_kernel<<<cB * cNH * (cL / 32), 256, 0, stream>>>(qkv16, o16);
    gemm_kernel<<<dim3(BL / 128, 256 / 64), 256, 0, stream>>>(
        o16, outwt + (size_t)l * 256 * 256, out_b + l * cH,
        h, h, nullptr, 256, 256, 0);
    ln_kernel<<<BL, 256, 0, stream>>>(h, ln2_w + l * cH, ln2_b + l * cH, a16);
    gemm_kernel<<<dim3(BL / 128, 1024 / 64), 256, 0, stream>>>(
        a16, ff1wt + (size_t)l * 1024 * 256, ff1_b + (size_t)l * 1024,
        nullptr, nullptr, ff16, 1024, 256, 1);
    gemm_kernel<<<dim3(BL / 128, 256 / 64), 256, 0, stream>>>(
        ff16, ff2wt + (size_t)l * 256 * 1024, ff2_b + l * cH,
        h, h, nullptr, 256, 1024, 0);
  }

  // --- pairwise head (LN hoisted through linearity) ---
  tofp16_kernel<<<BL, 256, 0, stream>>>(h, h16, BL * cH);
  stats_kernel<<<BL, 256, 0, stream>>>(h, muv, m2v);
  gemm_kernel<<<dim3(BL / 128, 256 / 64), 256, 0, stream>>>(
      h16, pw1a, nullptr, nullptr, Ubuf, nullptr, 256, 256, 0);
  gemm_kernel<<<dim3(BL / 128, 256 / 64), 256, 0, stream>>>(
      h16, pw1b, nullptr, nullptr, Vbuf, nullptr, 256, 256, 0);
  pair_kernel<<<BL, 256, 0, stream>>>(Ubuf, Vbuf, muv, m2v, zv, c0v, pb2, pw2t, outp);
  sym_kernel<<<(cB * cL * cL * 16) / 256, 256, 0, stream>>>(outp);
}
